// GKAN_Nodes_49469433315363
// MI455X (gfx1250) — matrix operand order repaired and checked by reading, not compiled, not run on hardware
//
#include <hip/hip_runtime.h>
#include <hip/hip_bf16.h>
#include <stdint.h>

typedef __attribute__((ext_vector_type(16))) _Float16 v16h;
typedef __attribute__((ext_vector_type(8)))  float    v8f;

#define N_NODES 20000
#define N_EDGES 320000
#define FEAT 128
#define NBF 7      // spline bases per input feature
#define KPF 8      // 1 silu + 7 bases -> augmented K per input feature
#define RT  4      // row tiles (of 16) per workgroup -> 64 rows

// ---------------- small utility kernels ----------------

__global__ void init_const_f(float* p, float v, int n) {
  int i = blockIdx.x * blockDim.x + threadIdx.x;
  if (i < n) p[i] = v;
}

__global__ void init_bias_rows(float* __restrict__ out, const float* __restrict__ bias,
                               int rows, int cols) {
  int i = blockIdx.x * blockDim.x + threadIdx.x;
  if (i < rows * cols) out[i] = bias[i % cols];
}

__global__ void deg_edges(const long long* __restrict__ ei, float* __restrict__ deg) {
  int e = blockIdx.x * blockDim.x + threadIdx.x;
  if (e < N_EDGES) atomicAdd(&deg[(int)ei[N_EDGES + e]], 1.0f);
}

__global__ void deg_to_dinv(float* d) {
  int i = blockIdx.x * blockDim.x + threadIdx.x;
  if (i < N_NODES) d[i] = rsqrtf(d[i]);   // deg >= 1 always (self loop)
}

// Pack base_w (O x in) and spline_w (O x in x 7) into Wt[o][i*8 + j] f16,
// row-major with K = in*8 contiguous per output row; rows o >= n_out zero-padded.
__global__ void pack_w(const float* __restrict__ base_w, const float* __restrict__ spline_w,
                       _Float16* __restrict__ Wt, int n_in, int n_out, int o_pad) {
  int K = n_in * KPF;
  int gid = blockIdx.x * blockDim.x + threadIdx.x;
  if (gid >= o_pad * K) return;
  int o = gid / K, k = gid - o * K;
  int i = k >> 3, j = k & 7;
  float v = 0.0f;
  if (o < n_out)
    v = (j == 0) ? base_w[o * n_in + i] : spline_w[(o * n_in + i) * NBF + (j - 1)];
  Wt[gid] = (_Float16)v;
}

// ---------------- spline + silu augmented features ----------------
// GRID = {-2.5, -2.0, ..., 2.5} (11 knots, h=0.5); Cox-de Boor order 3 -> 7 bases.
__device__ __forceinline__ void aug_features(float v, _Float16* dst) {
  float sig = 1.0f / (1.0f + expf(-v));
  float b[10];
#pragma unroll
  for (int j = 0; j < 10; ++j) {
    float gj = -2.5f + 0.5f * (float)j;
    b[j] = (v >= gj && v < gj + 0.5f) ? 1.0f : 0.0f;
  }
#pragma unroll
  for (int k = 1; k <= 3; ++k) {
    float inv = 2.0f / (float)k;  // 1 / (0.5*k): uniform-grid denominators
#pragma unroll
    for (int j = 0; j + k < 10; ++j) {
      float gj = -2.5f + 0.5f * (float)j;
      b[j] = (v - gj) * inv * b[j] + ((gj + 0.5f * (float)(k + 1)) - v) * inv * b[j + 1];
    }
  }
  dst[0] = (_Float16)(v * sig);
#pragma unroll
  for (int k = 0; k < NBF; ++k) dst[1 + k] = (_Float16)b[k];
}

// ---------------- fused KAN-linear GEMM via WMMA ----------------
// 64 rows per WG; 4 waves. Wave w owns col tiles {w, w+4, ...} (CT of them) and
// ALL 4 row tiles -> RT*CT accumulators; B-frags reused across row tiles.
// All RT A-frags are preloaded into distinct registers, then RT*CT WMMAs issue
// back-to-back (no per-tile WAR hazard NOPs).
template <int NSUP, int CT, int ACTW>
__global__ __launch_bounds__(128) void kan_gemm(
    const float* __restrict__ s0, const float* __restrict__ s1, const float* __restrict__ s2,
    const _Float16* __restrict__ Wt, float* __restrict__ Hout, int n_real)
{
  extern __shared__ _Float16 aug[];     // RT*16 x 1024 halves = 128 KB
  const int rowBase = blockIdx.x * (RT * 16);
  const int tid  = threadIdx.x;
  const int wave = tid >> 5, lane = tid & 31;
  const int lm = lane & 15, lh = lane >> 4;
  constexpr int Ktot = NSUP << 10;      // 1024 or 3072

  v8f acc[RT][CT] = {{}};

  // per-col-tile B row pointers (lane holds column tile*16+lm; K contiguous)
  const _Float16* bptr[CT];
#pragma unroll
  for (int i = 0; i < CT; ++i)
    bptr[i] = Wt + (size_t)((wave + i * 4) * 16 + lm) * Ktot + ((size_t)lh << 4);

  for (int s = 0; s < NSUP; ++s) {
    const float* src = (s == 0) ? s0 : ((s == 1) ? s1 : s2);
    // cooperatively build augmented panel: 64 rows x 128 feats -> 64 x 1024 halves
    for (int idx = tid; idx < RT * 16 * FEAT; idx += 128) {
      int r = idx >> 7, i = idx & 127;
      int row = rowBase + r;
      if (row >= N_NODES) row = N_NODES - 1;        // clamp (writeback guarded)
      float v = src[(size_t)row * FEAT + i];
      aug_features(v, &aug[(r << 10) + (i << 3)]);
    }
    __syncthreads();

    if (wave < ACTW) {                   // wave-uniform, hoisted out of K loop
#pragma unroll 2
      for (int kc = 0; kc < 32; ++kc) {
        const int kg = (s << 5) + kc;
        // B fragments: one aligned 32B load per col tile
        v16h bf[CT];
#pragma unroll
        for (int i = 0; i < CT; ++i)
          bf[i] = *(const v16h*)(bptr[i] + ((size_t)kg << 5));
        // preload ALL row-tile A fragments (contiguous k = 16*lh + s, matching B)
        union AF { v16h h; uint32_t u[8]; };
        AF af[RT];
#pragma unroll
        for (int r = 0; r < RT; ++r) {
          const uint32_t* lp =
              (const uint32_t*)&aug[((r * 16 + lm) << 10) + (kc << 5)];
#pragma unroll
          for (int p = 0; p < 8; ++p)
            af[r].u[p] = lp[p + 8 * lh];
        }
        // RT*CT independent WMMAs, back-to-back through the matrix pipe
#pragma unroll
        for (int r = 0; r < RT; ++r)
#pragma unroll
          for (int i = 0; i < CT; ++i)
            acc[r][i] = __builtin_amdgcn_wmma_f32_16x16x32_f16(
                false, af[r].h, false, bf[i], (short)0, acc[r][i], false, false);
      }
    }
    __syncthreads();
  }

  // write back C/D layout: lane col = lm, VGPR q -> row q + 8*lh
  if (wave < ACTW) {
#pragma unroll
    for (int r = 0; r < RT; ++r) {
#pragma unroll
      for (int i = 0; i < CT; ++i) {
        int col = (wave + i * 4) * 16 + lm;
        if (col < n_real) {
#pragma unroll
          for (int q = 0; q < 8; ++q) {
            int row = rowBase + r * 16 + q + 8 * lh;
            if (row < N_NODES)
              Hout[(size_t)row * n_real + col] = acc[r][i][q];
          }
        }
      }
    }
  }
}

// ---------------- edge aggregation (gather * norm, scatter-add) ----------------
__global__ void aggregate(const float* __restrict__ Hpre, const long long* __restrict__ ei,
                          const float* __restrict__ dinv, float* __restrict__ Hagg, int ncol) {
  int chunks = ncol >> 2;
  int gid = blockIdx.x * blockDim.x + threadIdx.x;
  int total = (N_EDGES + N_NODES) * chunks;
  if (gid >= total) return;
  int e = gid / chunks, c4 = (gid - e * chunks) << 2;
  int s, d;
  if (e < N_EDGES) { s = (int)ei[e]; d = (int)ei[N_EDGES + e]; }
  else             { s = d = e - N_EDGES; }          // self loop
  float nrm = dinv[s] * dinv[d];
  const float4 v = *(const float4*)(Hpre + (size_t)s * ncol + c4);
  float* o = Hagg + (size_t)d * ncol + c4;
  atomicAdd(o + 0, v.x * nrm);
  atomicAdd(o + 1, v.y * nrm);
  atomicAdd(o + 2, v.z * nrm);
  atomicAdd(o + 3, v.w * nrm);
}

// ---------------- batchnorm (biased var), in place, one block per channel ----------------
__global__ __launch_bounds__(256) void bn_inplace(float* __restrict__ h,
                                                  const float* __restrict__ gamma,
                                                  const float* __restrict__ beta) {
  const int c = blockIdx.x;
  __shared__ float ssum[256], ssq[256];
  __shared__ float sMean, sRstd;
  float sum = 0.f, sq = 0.f;
  for (int n = threadIdx.x; n < N_NODES; n += 256) {
    float v = h[(size_t)n * FEAT + c];
    sum += v; sq += v * v;
  }
  ssum[threadIdx.x] = sum; ssq[threadIdx.x] = sq;
  __syncthreads();
  for (int s = 128; s > 0; s >>= 1) {
    if (threadIdx.x < s) {
      ssum[threadIdx.x] += ssum[threadIdx.x + s];
      ssq[threadIdx.x]  += ssq[threadIdx.x + s];
    }
    __syncthreads();
  }
  if (threadIdx.x == 0) {
    float mean = ssum[0] / (float)N_NODES;
    float var  = ssq[0] / (float)N_NODES - mean * mean;
    sMean = mean;
    sRstd = rsqrtf(var + 1e-5f);
  }
  __syncthreads();
  float scale = sRstd * gamma[c];
  float shift = beta[c] - sMean * scale;
  for (int n = threadIdx.x; n < N_NODES; n += 256) {
    size_t off = (size_t)n * FEAT + c;
    h[off] = h[off] * scale + shift;
  }
}

// ---------------- host orchestration ----------------

extern "C" void kernel_launch(void* const* d_in, const int* in_sizes, int n_in,
                              void* d_out, int out_size, void* d_ws, size_t ws_size,
                              hipStream_t stream) {
  (void)in_sizes; (void)n_in; (void)out_size; (void)ws_size;
  const float*     x         = (const float*)d_in[0];
  const long long* ei        = (const long long*)d_in[1];   // int64 (2, 320000)
  const float*     base_w0   = (const float*)d_in[2];
  const float*     spline_w0 = (const float*)d_in[3];
  const float*     bias0     = (const float*)d_in[4];
  const float*     base_w1   = (const float*)d_in[5];
  const float*     spline_w1 = (const float*)d_in[6];
  const float*     bias1     = (const float*)d_in[7];
  const float*     base_wo   = (const float*)d_in[8];
  const float*     spline_wo = (const float*)d_in[9];
  const float*     bias_o    = (const float*)d_in[10];
  const float*     gamma     = (const float*)d_in[11];
  const float*     beta      = (const float*)d_in[12];
  float* out = (float*)d_out;

  // workspace carve-up (~32 MB)
  float* dinv = (float*)d_ws;                  // 20480 floats (padded)
  float* Hpre = dinv + 20480;                  // 20000*128 pre-aggregation buffer
  float* h1   = Hpre + 2560000;
  float* h2   = h1   + 2560000;
  _Float16* Wt0 = (_Float16*)(h2 + 2560000);   // 128 x 1024
  _Float16* Wt1 = Wt0 + 128 * 1024;            // 128 x 1024
  _Float16* WtO = Wt1 + 128 * 1024;            // 48 x 3072 (rows 40..47 zero)

  dim3 B(256);
  auto G = [](int n) { return dim3((n + 255) / 256); };

  // gcn_norm: deg (init 1 for self loop) -> dinv
  init_const_f<<<G(N_NODES), B, 0, stream>>>(dinv, 1.0f, N_NODES);
  deg_edges<<<G(N_EDGES), B, 0, stream>>>(ei, dinv);
  deg_to_dinv<<<G(N_NODES), B, 0, stream>>>(dinv);

  // pack weights
  pack_w<<<G(128 * 1024), B, 0, stream>>>(base_w0, spline_w0, Wt0, 128, 128, 128);
  pack_w<<<G(128 * 1024), B, 0, stream>>>(base_w1, spline_w1, Wt1, 128, 128, 128);
  pack_w<<<G(48 * 3072),  B, 0, stream>>>(base_wo, spline_wo, WtO, 384, 40, 48);

  const int gemmBlocks = (N_NODES + RT * 16 - 1) / (RT * 16);   // 313
  const size_t gemmLds = (size_t)RT * 16 * 1024 * sizeof(_Float16);  // 128 KB
  dim3 gemmBlk(128);

  // layer 0: KAN linear -> aggregate(+bias) -> BN
  kan_gemm<1, 2, 4><<<dim3(gemmBlocks), gemmBlk, gemmLds, stream>>>(
      x, nullptr, nullptr, Wt0, Hpre, 128);
  init_bias_rows<<<G(N_NODES * 128), B, 0, stream>>>(h1, bias0, N_NODES, 128);
  aggregate<<<G((N_EDGES + N_NODES) * 32), B, 0, stream>>>(Hpre, ei, dinv, h1, 128);
  bn_inplace<<<dim3(128), B, 0, stream>>>(h1, gamma, beta);

  // layer 1
  kan_gemm<1, 2, 4><<<dim3(gemmBlocks), gemmBlk, gemmLds, stream>>>(
      h1, nullptr, nullptr, Wt1, Hpre, 128);
  init_bias_rows<<<G(N_NODES * 128), B, 0, stream>>>(h2, bias1, N_NODES, 128);
  aggregate<<<G((N_EDGES + N_NODES) * 32), B, 0, stream>>>(Hpre, ei, dinv, h2, 128);
  bn_inplace<<<dim3(128), B, 0, stream>>>(h2, gamma, beta);

  // output layer on concat [x, h1, h2] (384 -> 40), col tiles 0..2 on waves 0..2
  kan_gemm<3, 1, 3><<<dim3(gemmBlocks), gemmBlk, gemmLds, stream>>>(
      x, h1, h2, WtO, Hpre, 40);
  init_bias_rows<<<G(N_NODES * 40), B, 0, stream>>>(out, bias_o, N_NODES, 40);
  aggregate<<<G((N_EDGES + N_NODES) * 10), B, 0, stream>>>(Hpre, ei, dinv, out, 40);
}
